// MultiQueryAttention_48395691491381
// MI455X (gfx1250) — matrix operand order repaired and checked by reading, not compiled, not run on hardware
//
#include <hip/hip_runtime.h>
#include <hip/hip_bf16.h>
#include <cstdint>

// Problem constants (B,T,C) = (2,2048,1024), HEADS=16, HD=64 (multi-query: 1 KV head)
#define B_      2
#define T_      2048
#define C_      1024
#define HEADS_  16
#define HD_     64
#define NQKV    1152          // C + 2*HD  (Q | K | V packed columns)
#define MTOT    (B_ * T_)     // 4096 rows

typedef __attribute__((ext_vector_type(16))) __bf16 v16bf;
typedef __attribute__((ext_vector_type(8)))  __bf16 v8bf;
typedef __attribute__((ext_vector_type(8)))  float  v8f;

static __device__ __forceinline__ unsigned short f2bf(float f) {
  unsigned u = __float_as_uint(f);
  u += 0x7fffu + ((u >> 16) & 1u);      // round-to-nearest-even
  return (unsigned short)(u >> 16);
}
static __device__ __forceinline__ float bf2f(unsigned short h) {
  return __uint_as_float(((unsigned)h) << 16);
}

// CDNA5 async global->LDS copy (16B per lane), tracked by ASYNCcnt.
// LDS operand = low 32 bits of the generic pointer (generic LDS address is
// {shared_aperture_hi32, lds_offset32}), same value the compiler uses for DS ops.
static __device__ __forceinline__ void async_g2l_b128(const void* gptr, void* lptr) {
  asm volatile("global_load_async_to_lds_b128 %0, %1, off"
               :: "v"((unsigned)(uintptr_t)lptr),
                  "v"((unsigned long long)(uintptr_t)gptr)
               : "memory");
}
static __device__ __forceinline__ void wait_async0() {
  asm volatile("s_wait_asynccnt 0x0" ::: "memory");
}

// ---------------------------------------------------------------- prep kernels
__global__ void cvt_f32_bf16(const float* __restrict__ src,
                             unsigned short* __restrict__ dst, int n) {
  int i = blockIdx.x * blockDim.x + threadIdx.x;
  if (i < n) dst[i] = f2bf(src[i]);
}

__global__ void pack_wqkv(const float* __restrict__ Wq, const float* __restrict__ Wk,
                          const float* __restrict__ Wv, const float* __restrict__ bq,
                          const float* __restrict__ bk, const float* __restrict__ bv,
                          unsigned short* __restrict__ wcat, float* __restrict__ bcat) {
  int i = blockIdx.x * blockDim.x + threadIdx.x;
  if (i < C_ * NQKV) {
    int k = i / NQKV, n = i - k * NQKV;
    float w;
    if (n < C_)            w = Wq[k * C_ + n];
    else if (n < C_ + HD_) w = Wk[k * HD_ + (n - C_)];
    else                   w = Wv[k * HD_ + (n - C_ - HD_)];
    wcat[i] = f2bf(w);
  }
  if (i < NQKV) {
    float b;
    if (i < C_)            b = bq[i];
    else if (i < C_ + HD_) b = bk[i - C_];
    else                   b = bv[i - C_ - HD_];
    bcat[i] = b;
  }
}

// ---------------------------------------------------------------- bf16 WMMA GEMM
// Out[M,N] = A[M,K] (bf16) @ W[K,N] (bf16) + bias[N].  Writes bf16 (outb) or f32 (outf).
// WG: 256 thr = 8 waves in 4(m) x 2(n).  WG tile 128x128, wave tile 32x64, K step 32.
// Per wave per K-step: 8 WMMAs from 2 A-frags + 4 B-frags.
__global__ __launch_bounds__(256) void gemm_bf16(
    const unsigned short* __restrict__ A, const unsigned short* __restrict__ W,
    const float* __restrict__ bias, unsigned short* __restrict__ outb,
    float* __restrict__ outf, int M, int N, int K) {
  __shared__ alignas(64) unsigned short As[128 * 32];   // A tile, natural [m][k]
  __shared__ alignas(64) unsigned short Bt[128 * 32];   // W tile, transposed [n][k]
  const int tid = threadIdx.x;
  const int lane = tid & 31, wid = tid >> 5;
  const int half = lane >> 4, l16 = lane & 15;
  const int wm = wid >> 1, wn = wid & 1;
  const int m0 = blockIdx.y * 128, n0 = blockIdx.x * 128;

  v8f acc[2][4] = {};
  for (int kk = 0; kk < K; kk += 32) {
    // stage A async: 128 rows x 32 cols (bf16) = 512 16B-chunks / 256 thr
#pragma unroll
    for (int c = 0; c < 2; ++c) {
      int chunk = tid + c * 256;
      int row = chunk >> 2, qo = (chunk & 3) * 8;
      async_g2l_b128(&A[(size_t)(m0 + row) * K + kk + qo], &As[row * 32 + qo]);
    }
    // stage W transposed: rows kk..kk+31, cols n0..n0+127 (coalesced read, scatter write)
#pragma unroll
    for (int c = 0; c < 2; ++c) {
      int chunk = tid + c * 256;
      int kl = chunk >> 4, nc = (chunk & 15) * 8;
      v8bf wrow = *(const v8bf*)&W[(size_t)(kk + kl) * N + n0 + nc];
#pragma unroll
      for (int j = 0; j < 8; ++j)
        Bt[(nc + j) * 32 + kl] = ((const unsigned short*)&wrow)[j];
    }
    if (kk + 32 < K)  // CDNA5 global_prefetch_b8 hint for next A tile
      __builtin_prefetch(&A[(size_t)(m0 + (tid >> 2)) * K + kk + 32], 0, 1);
    wait_async0();
    __syncthreads();

    // A-frag: lane row = M, elements = contiguous K run of 16 at half*16
    v16bf af[2];
#pragma unroll
    for (int sm = 0; sm < 2; ++sm) {
      int row = wm * 32 + sm * 16 + l16;
      v8bf lo = *(const v8bf*)&As[row * 32 + half * 16];
      v8bf hi = *(const v8bf*)&As[row * 32 + half * 16 + 8];
      af[sm] = __builtin_shufflevector(lo, hi, 0,1,2,3,4,5,6,7,8,9,10,11,12,13,14,15);
    }
    // B-frag: lane col = N, contiguous K run of 16 at half*16 (transposed tile)
    v16bf bfr[4];
#pragma unroll
    for (int sn = 0; sn < 4; ++sn) {
      int col = wn * 64 + sn * 16 + l16;
      bfr[sn] = *(const v16bf*)&Bt[col * 32 + half * 16];
    }
#pragma unroll
    for (int sm = 0; sm < 2; ++sm)
#pragma unroll
      for (int sn = 0; sn < 4; ++sn)
        acc[sm][sn] = __builtin_amdgcn_wmma_f32_16x16x32_bf16(
            false, af[sm], false, bfr[sn], (short)0, acc[sm][sn], false, false);
    __syncthreads();
  }
  // epilogue: D frag row = half*8 + r, col = l16
#pragma unroll
  for (int sm = 0; sm < 2; ++sm)
#pragma unroll
    for (int sn = 0; sn < 4; ++sn)
#pragma unroll
      for (int r = 0; r < 8; ++r) {
        int row = m0 + wm * 32 + sm * 16 + half * 8 + r;
        int col = n0 + wn * 64 + sn * 16 + l16;
        float v = acc[sm][sn][r] + bias[col];
        if (outf) outf[(size_t)row * N + col] = v;
        else      outb[(size_t)row * N + col] = f2bf(v);
      }
}

// ---------------------------------------------------------------- RoPE + split
// qkv[MTOT][1152] bf16 -> q_bf[MTOT][1024] (roped), k_bf[MTOT][64] (roped), v_bf[MTOT][64]
__global__ void rope_split(const unsigned short* __restrict__ qkv,
                           unsigned short* __restrict__ qb,
                           unsigned short* __restrict__ kb,
                           unsigned short* __restrict__ vb) {
  const float NEG_LN1E4_OVER_32 = -0.28782313662425575f;  // -ln(10000)/32
  int idx = blockIdx.x * blockDim.x + threadIdx.x;
  const int NQ = MTOT * HEADS_ * 32;
  const int NK = MTOT * 32;
  const int NV = MTOT * HD_;
  if (idx < NQ) {
    int i = idx & 31;
    int h = (idx >> 5) & (HEADS_ - 1);
    int bt = idx >> 9;
    int t = bt & (T_ - 1);
    const unsigned short* src = qkv + (size_t)bt * NQKV + h * HD_;
    float q1 = bf2f(src[i]), q2 = bf2f(src[i + 32]);
    float ang = (float)t * __expf((float)i * NEG_LN1E4_OVER_32);
    float s, c; __sincosf(ang, &s, &c);
    unsigned short* dst = qb + (size_t)bt * C_ + h * HD_;
    dst[i]      = f2bf(q1 * c + q2 * s);
    dst[i + 32] = f2bf(q1 * s - q2 * c);
    return;
  }
  idx -= NQ;
  if (idx < NK) {
    int i = idx & 31;
    int bt = idx >> 5;
    int t = bt & (T_ - 1);
    const unsigned short* src = qkv + (size_t)bt * NQKV + C_;
    float k1 = bf2f(src[i]), k2 = bf2f(src[i + 32]);
    float ang = (float)t * __expf((float)i * NEG_LN1E4_OVER_32);
    float s, c; __sincosf(ang, &s, &c);
    unsigned short* dst = kb + (size_t)bt * HD_;
    dst[i]      = f2bf(k1 * c + k2 * s);
    dst[i + 32] = f2bf(k1 * s - k2 * c);
    return;
  }
  idx -= NK;
  if (idx < NV) {
    int j = idx & 63;
    int bt = idx >> 6;
    vb[(size_t)bt * HD_ + j] = qkv[(size_t)bt * NQKV + C_ + HD_ + j];
  }
}

// ---------------------------------------------------------------- flash MQA attention
// Grid (T/64, HEADS, B); 128 threads = 4 waves, each wave owns 16 query rows.
// Per 32-key block: S = 4 WMMAs, online softmax in D-frag layout with shfl_xor row
// reductions, P via LDS relayout, O += 4 WMMAs.  K tile staged via async-to-LDS.
__global__ __launch_bounds__(128) void mqa_attn(
    const unsigned short* __restrict__ qb, const unsigned short* __restrict__ kb,
    const unsigned short* __restrict__ vb, unsigned short* __restrict__ ob) {
  __shared__ alignas(64) unsigned short Ks[32 * 64];       // K block, natural [key][hd]
  __shared__ alignas(64) unsigned short Vt[64 * 32];       // V block, transposed [hd][key]
  __shared__ alignas(64) unsigned short Ps[4 * 16 * 32];   // per-wave P tiles
  const int tid = threadIdx.x, lane = tid & 31, wid = tid >> 5;
  const int half = lane >> 4, l16 = lane & 15;
  const int q0 = blockIdx.x * 64;
  const int h  = blockIdx.y;
  const int b  = blockIdx.z;
  const int qrow = q0 + wid * 16 + l16;

  // persistent Q fragments (hd 0-31, 32-63)
  const unsigned short* qptr = qb + ((size_t)b * T_ + qrow) * C_ + h * HD_;
  v16bf qf[2];
#pragma unroll
  for (int f = 0; f < 2; ++f) {
    v8bf lo = *(const v8bf*)&qptr[f * 32 + half * 16];
    v8bf hi = *(const v8bf*)&qptr[f * 32 + half * 16 + 8];
    qf[f] = __builtin_shufflevector(lo, hi, 0,1,2,3,4,5,6,7,8,9,10,11,12,13,14,15);
  }

  v8f accO[4] = {};
  float mstat[8], lstat[8];
#pragma unroll
  for (int r = 0; r < 8; ++r) { mstat[r] = -1e30f; lstat[r] = 0.f; }
  unsigned short* Pw = Ps + wid * 16 * 32;

  const int nkb = (q0 + 64) >> 5;  // causal: key blocks of 32 up to q0+63
  for (int kbI = 0; kbI < nkb; ++kbI) {
    const int kbase = kbI * 32;
    // stage K natural via CDNA5 async-to-LDS (256 16B-chunks / 128 thr)
#pragma unroll
    for (int c = 0; c < 2; ++c) {
      int chunk = tid + c * 128;
      int row = chunk >> 3, qo = (chunk & 7) * 8;
      async_g2l_b128(&kb[((size_t)b * T_ + kbase + row) * HD_ + qo],
                     &Ks[row * 64 + qo]);
    }
    // stage V transposed (coalesced read, scatter write)
#pragma unroll
    for (int c = 0; c < 2; ++c) {
      int chunk = tid + c * 128;
      int key = chunk >> 3, nc = (chunk & 7) * 8;
      v8bf vrow = *(const v8bf*)&vb[((size_t)b * T_ + kbase + key) * HD_ + nc];
#pragma unroll
      for (int j = 0; j < 8; ++j)
        Vt[(nc + j) * 32 + key] = ((const unsigned short*)&vrow)[j];
    }
    wait_async0();
    __syncthreads();

    // S = Q . K^T for two 16-key subtiles
    v8f s[2];
#pragma unroll
    for (int kt = 0; kt < 2; ++kt) {
      v8f z = {};
#pragma unroll
      for (int f = 0; f < 2; ++f) {
        v16bf kf = *(const v16bf*)&Ks[(kt * 16 + l16) * 64 + f * 32 + half * 16];
        z = __builtin_amdgcn_wmma_f32_16x16x32_bf16(false, qf[f], false, kf,
                                                    (short)0, z, false, false);
      }
      s[kt] = z;
    }
    // scale + causal mask (D-frag: row = half*8+r, col = l16)
#pragma unroll
    for (int kt = 0; kt < 2; ++kt)
#pragma unroll
      for (int r = 0; r < 8; ++r) {
        int row = q0 + wid * 16 + half * 8 + r;
        int key = kbase + kt * 16 + l16;
        float v = s[kt][r] * 0.125f;   // 1/sqrt(64)
        s[kt][r] = (key > row) ? -1e30f : v;
      }
    // online softmax update over 32-key block
    float p0[8], p1[8], scl[8];
#pragma unroll
    for (int r = 0; r < 8; ++r) {
      float tmax = fmaxf(s[0][r], s[1][r]);
#pragma unroll
      for (int off = 1; off < 16; off <<= 1)
        tmax = fmaxf(tmax, __shfl_xor(tmax, off, 32));
      float mnew = fmaxf(mstat[r], tmax);
      scl[r] = __expf(mstat[r] - mnew);
      p0[r] = __expf(s[0][r] - mnew);
      p1[r] = __expf(s[1][r] - mnew);
      float rs = p0[r] + p1[r];
#pragma unroll
      for (int off = 1; off < 16; off <<= 1)
        rs += __shfl_xor(rs, off, 32);
      lstat[r] = lstat[r] * scl[r] + rs;
      mstat[r] = mnew;
    }
#pragma unroll
    for (int j = 0; j < 4; ++j)
#pragma unroll
      for (int r = 0; r < 8; ++r)
        accO[j][r] *= scl[r];
    // write P (bf16) to this wave's LDS tile: 16 rows x 32 keys
#pragma unroll
    for (int r = 0; r < 8; ++r) {
      int row = half * 8 + r;
      Pw[row * 32 + l16]      = f2bf(p0[r]);
      Pw[row * 32 + 16 + l16] = f2bf(p1[r]);
    }
    __syncthreads();
    // reload P as A-frag; V frag from transposed tile; O += P.V
    v8bf plo = *(const v8bf*)&Pw[l16 * 32 + half * 16];
    v8bf phi = *(const v8bf*)&Pw[l16 * 32 + half * 16 + 8];
    v16bf pf = __builtin_shufflevector(plo, phi, 0,1,2,3,4,5,6,7,8,9,10,11,12,13,14,15);
#pragma unroll
    for (int j = 0; j < 4; ++j) {
      v16bf vf = *(const v16bf*)&Vt[(j * 16 + l16) * 32 + half * 16];
      accO[j] = __builtin_amdgcn_wmma_f32_16x16x32_bf16(false, pf, false, vf,
                                                        (short)0, accO[j], false, false);
    }
    __syncthreads();
  }
  // normalize and store bf16 attention output (B,T,C layout)
#pragma unroll
  for (int j = 0; j < 4; ++j)
#pragma unroll
    for (int r = 0; r < 8; ++r) {
      int row = q0 + wid * 16 + half * 8 + r;
      int hd = j * 16 + l16;
      float v = accO[j][r] / lstat[r];
      ob[((size_t)b * T_ + row) * C_ + h * HD_ + hd] = f2bf(v);
    }
}

// ---------------------------------------------------------------- host launcher
extern "C" void kernel_launch(void* const* d_in, const int* in_sizes, int n_in,
                              void* d_out, int out_size, void* d_ws, size_t ws_size,
                              hipStream_t stream) {
  const float* x  = (const float*)d_in[0];
  const float* Wq = (const float*)d_in[1];
  const float* bq = (const float*)d_in[2];
  const float* Wk = (const float*)d_in[3];
  const float* bk = (const float*)d_in[4];
  const float* Wv = (const float*)d_in[5];
  const float* bv = (const float*)d_in[6];
  const float* Wo = (const float*)d_in[7];
  const float* bo = (const float*)d_in[8];
  float* out = (float*)d_out;

  char* ws = (char*)d_ws;
  size_t off = 0;
  auto take = [&](size_t bytes) { char* p = ws + off; off = (off + bytes + 255) & ~(size_t)255; return p; };
  unsigned short* x_bf  = (unsigned short*)take((size_t)MTOT * C_ * 2);
  unsigned short* wcat  = (unsigned short*)take((size_t)C_ * NQKV * 2);
  float*          bcat  = (float*)take((size_t)NQKV * 4);
  unsigned short* qkv   = (unsigned short*)take((size_t)MTOT * NQKV * 2);
  unsigned short* q_bf  = (unsigned short*)take((size_t)MTOT * C_ * 2);
  unsigned short* k_bf  = (unsigned short*)take((size_t)MTOT * HD_ * 2);
  unsigned short* v_bf  = (unsigned short*)take((size_t)MTOT * HD_ * 2);
  unsigned short* attn  = (unsigned short*)take((size_t)MTOT * C_ * 2);
  unsigned short* wo_bf = (unsigned short*)take((size_t)C_ * C_ * 2);
  (void)ws_size; (void)n_in; (void)in_sizes; (void)out_size;

  // 1) convert x and Wo to bf16; pack Wq|Wk|Wv + biases
  cvt_f32_bf16<<<(MTOT * C_ + 255) / 256, 256, 0, stream>>>(x, x_bf, MTOT * C_);
  cvt_f32_bf16<<<(C_ * C_ + 255) / 256, 256, 0, stream>>>(Wo, wo_bf, C_ * C_);
  pack_wqkv<<<(C_ * NQKV + 255) / 256, 256, 0, stream>>>(Wq, Wk, Wv, bq, bk, bv, wcat, bcat);

  // 2) fused QKV projection: [4096,1024] @ [1024,1152] + bias -> bf16
  gemm_bf16<<<dim3(NQKV / 128, MTOT / 128), 256, 0, stream>>>(
      x_bf, wcat, bcat, qkv, nullptr, MTOT, NQKV, C_);

  // 3) RoPE + split into q/k/v buffers
  rope_split<<<(MTOT * (HEADS_ * 32 + 32 + HD_) + 255) / 256, 256, 0, stream>>>(
      qkv, q_bf, k_bf, v_bf);

  // 4) causal flash MQA attention
  mqa_attn<<<dim3(T_ / 64, HEADS_, B_), 128, 0, stream>>>(q_bf, k_bf, v_bf, attn);

  // 5) output projection: [4096,1024] @ [1024,1024] + bo -> f32 d_out
  gemm_bf16<<<dim3(C_ / 128, MTOT / 128), 256, 0, stream>>>(
      attn, wo_bf, bo, nullptr, out, MTOT, C_, C_);
}
